// MultiHeadAttention_69063074120143
// MI455X (gfx1250) — matrix operand order repaired and checked by reading, not compiled, not run on hardware
//
#include <hip/hip_runtime.h>
#include <hip/hip_bf16.h>

// ---------------------------------------------------------------------------
// Multi-head attention for MI455X (gfx1250) — all matmuls on
// v_wmma_f32_16x16x32_f16; staging pipelined; attention K tiles staged with
// gfx1250 async global->LDS DMA (ASYNCcnt) when the toolchain supports it.
// ---------------------------------------------------------------------------

typedef __attribute__((ext_vector_type(16))) _Float16 v16h;
typedef __attribute__((ext_vector_type(8)))  _Float16 v8h;
typedef __attribute__((ext_vector_type(8)))  float    v8f;
typedef __attribute__((ext_vector_type(4)))  float    v4f;
typedef __attribute__((ext_vector_type(4)))  int      v4i;

constexpr int B_  = 4;
constexpr int S_  = 2048;
constexpr int DIN = 1024;
constexpr int H_  = 16;
constexpr int DH  = 64;

constexpr int WAVES = 8;            // waves per block (wave32)
constexpr int BLK   = WAVES * 32;   // 256 threads
constexpr int MT    = 16;           // M rows per wave tile
constexpr int MSUP  = WAVES * MT;   // 128 rows per block
constexpr int KB    = 32;           // K-step / key-block (matches WMMA K=32)

#if defined(__gfx1250__) && __has_builtin(__builtin_amdgcn_global_load_async_to_lds_b128)
#define HAVE_ASYNC_LDS 1
#endif

__device__ __forceinline__ v8f wmma_f16(const v16h& a, const v16h& b, const v8f& c) {
  return __builtin_amdgcn_wmma_f32_16x16x32_f16(false, a, false, b, (short)0, c, false, false);
}

__device__ __forceinline__ v16h join(v8h lo, v8h hi) {
  return __builtin_shufflevector(lo, hi, 0, 1, 2, 3, 4, 5, 6, 7,
                                 8, 9, 10, 11, 12, 13, 14, 15);
}
__device__ __forceinline__ v8h ld8h(const _Float16* p) {     // 16B load
  return *(const v8h*)p;
}
__device__ __forceinline__ v8h cvt8(const float* p) {        // 2x16B load + cvt
  v4f a = *(const v4f*)p;
  v4f b = *(const v4f*)(p + 4);
  v8h r;
#pragma unroll
  for (int i = 0; i < 4; ++i) { r[i] = (_Float16)a[i]; r[4 + i] = (_Float16)b[i]; }
  return r;
}

// Async 16B global -> LDS copy (per-lane), tracked by ASYNCcnt.
// Builtin signature (from hipcc diagnostic): (v4i AS1*, v4i AS3*, imm, imm).
__device__ __forceinline__ void async_copy16(const _Float16* g, _Float16* l) {
#if HAVE_ASYNC_LDS
  __builtin_amdgcn_global_load_async_to_lds_b128(
      (__attribute__((address_space(1))) v4i*)(g),
      (__attribute__((address_space(3))) v4i*)(l), 0, 0);
#else
  *(v8h*)l = *(const v8h*)g;
#endif
}
__device__ __forceinline__ void wait_async0() {
#if HAVE_ASYNC_LDS
#if __has_builtin(__builtin_amdgcn_s_wait_asynccnt)
  __builtin_amdgcn_s_wait_asynccnt(0);
#else
  asm volatile("s_wait_asynccnt 0x0" ::: "memory");
#endif
#endif
}

// ---------------------------------------------------------------------------
// Per-head projection: Out[b,h,s,d] = (X[b,s,:] @ W[h,:,:] + bias[h,:]) * scale
// X: [B*S, DIN] f32, W: [H, DIN, DH] f32, Out: [B,H,S,DH] f16
// Pipelined: scatter tile k, barrier, issue loads for tile k+1, WMMA on tile k.
// ---------------------------------------------------------------------------
__global__ __launch_bounds__(BLK) void proj_kernel(const float* __restrict__ X,
                                                   const float* __restrict__ W,
                                                   const float* __restrict__ bias,
                                                   _Float16* __restrict__ Out,
                                                   float scale) {
  // W k-tile pre-swizzled into B-fragment layout: [n-chunk][lane][element]
  __shared__ __align__(16) _Float16 WsF[4][32][16];

  const int tid  = threadIdx.x;
  const int wave = tid >> 5;
  const int lane = tid & 31;
  const int lm   = lane & 15;
  const int lh   = lane >> 4;
  const int h    = blockIdx.y;
  const int m0   = blockIdx.x * MSUP + wave * MT;
  const int row  = m0 + lm;                       // A-matrix row for this lane
  const float* Wh = W + (size_t)h * DIN * DH;

  // staging geometry: each thread owns 8 contiguous cols of one k-row
  const int sr  = tid >> 3;              // k row 0..31
  const int sc0 = (tid & 7) * 8;         // col 0,8,...,56
  const int scg = sc0 >> 4;              // n-chunk (constant over the 8 cols)
  const int seb = sr & 15;               // element index
  const int slb = (sr >> 4) << 4;        // lane-half base

  v8f acc[4];
#pragma unroll
  for (int c = 0; c < 4; ++c)
#pragma unroll
    for (int j = 0; j < 8; ++j) acc[c][j] = 0.0f;

  v8h wreg = cvt8(&Wh[(size_t)sr * DH + sc0]);    // tile 0 in registers

  for (int k0 = 0; k0 < DIN; k0 += KB) {
    // publish current W tile (reads of previous tile finished at prior barrier)
#pragma unroll
    for (int t = 0; t < 8; ++t) WsF[scg][slb | ((sc0 + t) & 15)][seb] = wreg[t];
    __syncthreads();

    if (k0 + KB < DIN)                            // overlap next load with WMMAs
      wreg = cvt8(&Wh[(size_t)(k0 + KB + sr) * DH + sc0]);

    // A fragment: two contiguous 8-half chunks (b128 global loads + cvt)
    v16h a = join(cvt8(&X[(size_t)row * DIN + k0 + lh * 16]),
                  cvt8(&X[(size_t)row * DIN + k0 + lh * 16 + 8]));

#pragma unroll
    for (int c = 0; c < 4; ++c) {
      const _Float16* bp = &WsF[c][lane][0];
      v16h b = join(ld8h(bp), ld8h(bp + 8));      // 2x ds_load_b128
      acc[c] = wmma_f16(a, b, acc[c]);
    }
    __syncthreads();
  }

  // Epilogue: +bias, *scale, store f16 to [B,H,S,DH] (coalesced across lm)
#pragma unroll
  for (int c = 0; c < 4; ++c) {
    int d = c * 16 + lm;
    float bv = bias[h * DH + d];
#pragma unroll
    for (int j = 0; j < 8; ++j) {
      int orow = m0 + j + lh * 8;                 // C-layout row
      int bb = orow >> 11;                        // / S_
      int ss = orow & (S_ - 1);
      Out[(((size_t)bb * H_ + h) * S_ + ss) * DH + d] =
          (_Float16)((acc[c][j] + bv) * scale);
    }
  }
}

// ---------------------------------------------------------------------------
// Flash-attention per (b,h): Q pre-scaled by 1/sqrt(DH).
// Qb/Kb/Vb: [B,H,S,DH] f16.  Ao: [B,S,H*DH] f16.
// K tiles: double-buffered async global->LDS DMA overlapping compute.
// V tiles: register-staged + fragment-swizzled scatter, pipelined.
// ---------------------------------------------------------------------------
__global__ __launch_bounds__(BLK) void attn_kernel(const _Float16* __restrict__ Qb,
                                                   const _Float16* __restrict__ Kb,
                                                   const _Float16* __restrict__ Vb,
                                                   _Float16* __restrict__ Ao) {
  __shared__ __align__(16) _Float16 Ks[2][KB][DH + 8];     // double-buffered (async DMA)
  __shared__ __align__(16) _Float16 VsF[4][32][16];        // V in B-fragment layout
  __shared__ __align__(16) _Float16 Ps[WAVES][16][KB + 8]; // pitch 40 halfs = 80B

  const int tid  = threadIdx.x;
  const int wave = tid >> 5;
  const int lane = tid & 31;
  const int lm   = lane & 15;
  const int lh   = lane >> 4;
  const int bh   = blockIdx.y;                    // b*H + h
  const size_t base = (size_t)bh * S_ * DH;
  const int m0   = blockIdx.x * MSUP + wave * MT; // query row tile within S

  // staging geometry
  const int sr  = tid >> 3;
  const int sc0 = (tid & 7) * 8;
  const int scg = sc0 >> 4;
  const int seb = sr & 15;
  const int slb = (sr >> 4) << 4;
  const size_t sgo = (size_t)sr * DH + sc0;       // offset within a tile's rows

  // Q fragments held in registers whole kernel (b128 global loads)
  v16h qf[2];
#pragma unroll
  for (int f = 0; f < 2; ++f) {
    const _Float16* qp = &Qb[base + (size_t)(m0 + lm) * DH + f * 32 + lh * 16];
    qf[f] = join(ld8h(qp), ld8h(qp + 8));
  }

  v8f o[4];
  float rm[8], rs[8];
#pragma unroll
  for (int c = 0; c < 4; ++c)
#pragma unroll
    for (int j = 0; j < 8; ++j) o[c][j] = 0.0f;
#pragma unroll
  for (int j = 0; j < 8; ++j) { rm[j] = -1e30f; rs[j] = 0.0f; }

  // Prologue: async-DMA K tile 0, V tile 0 into registers
  async_copy16(&Kb[base + sgo], &Ks[0][sr][sc0]);
  v8h vv = ld8h(&Vb[base + sgo]);

  const int nkb = S_ / KB;  // 64 key blocks
  for (int kb = 0; kb < nkb; ++kb) {
    const int cur = kb & 1;

    // publish V tile kb (fragment-swizzled scatter); join K DMA for tile kb
#pragma unroll
    for (int t = 0; t < 8; ++t) VsF[scg][slb | ((sc0 + t) & 15)][seb] = vv[t];
    wait_async0();
    __syncthreads();

    if (kb + 1 < nkb) {  // overlap next tile's traffic with this tile's WMMAs
      size_t g = base + (size_t)(kb + 1) * KB * DH + sgo;
      async_copy16(&Kb[g], &Ks[cur ^ 1][sr][sc0]);
      vv = ld8h(&Vb[g]);
    }

    // scores S = Q . K^T  (16 q-rows x 32 keys): two 16x16 C frags
    v8f s0, s1;
#pragma unroll
    for (int j = 0; j < 8; ++j) { s0[j] = 0.0f; s1[j] = 0.0f; }
#pragma unroll
    for (int f = 0; f < 2; ++f) {
      const _Float16* k0p = &Ks[cur][lm][f * 32 + lh * 16];       // contiguous 32B
      const _Float16* k1p = &Ks[cur][16 + lm][f * 32 + lh * 16];
      v16h b0 = join(ld8h(k0p), ld8h(k0p + 8));
      v16h b1 = join(ld8h(k1p), ld8h(k1p + 8));
      s0 = wmma_f16(qf[f], b0, s0);
      s1 = wmma_f16(qf[f], b1, s1);
    }

    // Online softmax: rows r = j + lh*8, reduce across the 16-lane N-group
#pragma unroll
    for (int j = 0; j < 8; ++j) {
      float t = fmaxf(s0[j], s1[j]);
#pragma unroll
      for (int m = 1; m < 16; m <<= 1) t = fmaxf(t, __shfl_xor(t, m, 32));
      float mnew  = fmaxf(rm[j], t);
      float alpha = __expf(rm[j] - mnew);
      float p0 = __expf(s0[j] - mnew);
      float p1 = __expf(s1[j] - mnew);
      float psum = p0 + p1;
#pragma unroll
      for (int m = 1; m < 16; m <<= 1) psum += __shfl_xor(psum, m, 32);
      rs[j] = rs[j] * alpha + psum;
      rm[j] = mnew;
#pragma unroll
      for (int c = 0; c < 4; ++c) o[c][j] *= alpha;
      int r = j + lh * 8;                          // C-layout -> LDS re-shape
      Ps[wave][r][lm]      = (_Float16)p0;
      Ps[wave][r][16 + lm] = (_Float16)p1;
    }

    // P as A fragment (same-wave LDS ops are in order): 2x ds_load_b128
    const _Float16* pp = &Ps[wave][lm][lh * 16];
    v16h pa = join(ld8h(pp), ld8h(pp + 8));

    // O += P . V  (V fragments pre-swizzled: 2x ds_load_b128 each)
#pragma unroll
    for (int c = 0; c < 4; ++c) {
      const _Float16* vp = &VsF[c][lane][0];
      v16h vb = join(ld8h(vp), ld8h(vp + 8));
      o[c] = wmma_f16(pa, vb, o[c]);
    }
    __syncthreads();   // all reads of Ks[cur]/VsF done before next publish
  }

  // Normalize and store heads-concatenated f16 [B,S,H*DH]
  const int b = bh >> 4;
  const int h = bh & 15;
#pragma unroll
  for (int j = 0; j < 8; ++j) {
    float inv = 1.0f / rs[j];
    int qrow = m0 + j + lh * 8;
#pragma unroll
    for (int c = 0; c < 4; ++c) {
      int d = c * 16 + lm;
      Ao[((size_t)b * S_ + qrow) * (H_ * DH) + h * DH + d] = (_Float16)(o[c][j] * inv);
    }
  }
}

// ---------------------------------------------------------------------------
// Output projection: Out[m, :] = Ao[m, :] @ Wo + bo   (M=8192, K=1024, N=1024)
// ---------------------------------------------------------------------------
__global__ __launch_bounds__(BLK) void outproj_kernel(const _Float16* __restrict__ Ao,
                                                      const float* __restrict__ Wo,
                                                      const float* __restrict__ bo,
                                                      float* __restrict__ Out) {
  __shared__ __align__(16) _Float16 WsF[4][32][16];

  const int tid  = threadIdx.x;
  const int wave = tid >> 5;
  const int lane = tid & 31;
  const int lm   = lane & 15;
  const int lh   = lane >> 4;
  const int n0   = blockIdx.y * 64;
  const int m0   = blockIdx.x * MSUP + wave * MT;
  const int row  = m0 + lm;

  const int sr  = tid >> 3;
  const int sc0 = (tid & 7) * 8;
  const int scg = sc0 >> 4;
  const int seb = sr & 15;
  const int slb = (sr >> 4) << 4;

  v8f acc[4];
#pragma unroll
  for (int c = 0; c < 4; ++c)
#pragma unroll
    for (int j = 0; j < 8; ++j) acc[c][j] = 0.0f;

  v8h wreg = cvt8(&Wo[(size_t)sr * DIN + n0 + sc0]);   // tile 0

  for (int k0 = 0; k0 < DIN; k0 += KB) {
#pragma unroll
    for (int t = 0; t < 8; ++t) WsF[scg][slb | ((sc0 + t) & 15)][seb] = wreg[t];
    __syncthreads();

    if (k0 + KB < DIN)
      wreg = cvt8(&Wo[(size_t)(k0 + KB + sr) * DIN + n0 + sc0]);

    const _Float16* ap = &Ao[(size_t)row * DIN + k0 + lh * 16];
    v16h a = join(ld8h(ap), ld8h(ap + 8));

#pragma unroll
    for (int c = 0; c < 4; ++c) {
      const _Float16* bp = &WsF[c][lane][0];
      v16h b = join(ld8h(bp), ld8h(bp + 8));
      acc[c] = wmma_f16(a, b, acc[c]);
    }
    __syncthreads();
  }

#pragma unroll
  for (int c = 0; c < 4; ++c) {
    int n = n0 + c * 16 + lm;
    float bv = bo[n];
#pragma unroll
    for (int j = 0; j < 8; ++j) {
      int orow = m0 + j + lh * 8;
      Out[(size_t)orow * DIN + n] = acc[c][j] + bv;
    }
  }
}

// ---------------------------------------------------------------------------
extern "C" void kernel_launch(void* const* d_in, const int* in_sizes, int n_in,
                              void* d_out, int out_size, void* d_ws, size_t ws_size,
                              hipStream_t stream) {
  (void)in_sizes; (void)n_in; (void)out_size; (void)ws_size;

  const float* query = (const float*)d_in[0];
  const float* key   = (const float*)d_in[1];
  const float* value = (const float*)d_in[2];
  const float* Wq    = (const float*)d_in[3];
  const float* bq    = (const float*)d_in[4];
  const float* Wk    = (const float*)d_in[5];
  const float* bk    = (const float*)d_in[6];
  const float* Wv    = (const float*)d_in[7];
  const float* bv    = (const float*)d_in[8];
  const float* Wo    = (const float*)d_in[9];
  const float* bo    = (const float*)d_in[10];
  float* out = (float*)d_out;

  const size_t n = (size_t)B_ * H_ * S_ * DH;   // 8,388,608 elems per buffer
  _Float16* Qb = (_Float16*)d_ws;
  _Float16* Kb = Qb + n;
  _Float16* Vb = Kb + n;
  _Float16* Ao = Vb + n;                        // [B,S,H*DH] f16

  dim3 blk(BLK);

  dim3 gproj(B_ * S_ / MSUP, H_);               // 64 x 16
  proj_kernel<<<gproj, blk, 0, stream>>>(query, Wq, bq, Qb, 0.125f); // 1/sqrt(64)
  proj_kernel<<<gproj, blk, 0, stream>>>(key,   Wk, bk, Kb, 1.0f);
  proj_kernel<<<gproj, blk, 0, stream>>>(value, Wv, bv, Vb, 1.0f);

  dim3 gattn(S_ / MSUP, B_ * H_);               // 16 x 64
  attn_kernel<<<gattn, blk, 0, stream>>>(Qb, Kb, Vb, Ao);

  dim3 gout(B_ * S_ / MSUP, DIN / 64);          // 64 x 16
  outproj_kernel<<<gout, blk, 0, stream>>>(Ao, Wo, bo, out);
}
